// EGCLCrowdBase_17532056502807
// MI455X (gfx1250) — matrix operand order repaired and checked by reading, not compiled, not run on hardware
//
#include <hip/hip_runtime.h>
#include <hip/hip_bf16.h>
#include <math.h>

// EGCL crowd model, fully fused: 1 workgroup per batch element.
// BS=1024, N=64 agents, K=6 neighbors (padded to 8), D=128, OBS=21.
// All GEMMs via v_wmma_f32_16x16x32_f16 (f16 operands, f32 accumulate).

#define NAG     64
#define KNN     6
#define KP      8
#define DIM     128
#define OBSF    21
#define XROW    (NAG*OBSF + 2*NAG)   // 1472
#define THREADS 256
#define NWAVE   8
#define NBATCH  1024

typedef _Float16 h8_t __attribute__((ext_vector_type(8)));
typedef _Float16 v16h __attribute__((ext_vector_type(16)));
typedef float    v8f  __attribute__((ext_vector_type(8)));

struct LayerP {
  const float *fe0w,*fe0b,*fe1w,*fe1b,*fx0w,*fx0b,*fx1w,*fx1b,*fh0w,*fh0b,*fh1w,*fh1b;
};
struct Args {
  const float* x;
  const float *e0w,*e0b,*e1w,*e1b;
  LayerP L[3];
  float* out;
};

// ---- dynamic LDS layout (~297 KB, fits CDNA5 320KB/WGP) ----
struct alignas(16) Smem {
  float     h32[NAG*DIM];          // 32 KB  h_st f32 (residual + output)
  _Float16  h16[NAG*DIM];          // 16 KB  f16 mirror (WMMA A operand)
  _Float16  hm [NAG*2*DIM];        // 32 KB  [h_st | m_i] f16 for fh0
  _Float16  WT0[128*288];          // 72 KB  fe0^T (288=K pad) / fh0^T (256 used)
  _Float16  WT1[128*128];          // 32 KB  fe1^T / fh1^T / enc1^T
  _Float16  WT2[128*128];          // 32 KB  fx0^T / enc0^T (32 used)
  _Float16  stageS[NWAVE*16*DIM];  // 32 KB  per-wave stage; reused as fh0-out [64][128]
  _Float16  stageM[NWAVE*16*DIM];  // 32 KB  per-wave m tile; reused as encoder stage
  _Float16  featS[NAG*32];         //  4 KB  encoder input (17 -> pad 32)
  float     biasA[128], biasB[128], biasC[128];
  float     fx1w[128];
  float     fx1b;
  float     posx[NAG], posy[NAG];
  int       nbrA[NAG*KP];
  float     maskA[NAG*KP];
  float     relxA[NAG*KP], relyA[NAG*KP], distA[NAG*KP];
  float     nnumA[NAG];
  float     validf[NAG];
};

__device__ __forceinline__ float silu_f(float v) { return v / (1.0f + __expf(-v)); }

__device__ __forceinline__ v8f wmma16(v16h a, v16h b, v8f c) {
  // (neg_a, A, neg_b, B, c_mod, C, reuse_a, reuse_b)
  return __builtin_amdgcn_wmma_f32_16x16x32_f16(false, a, false, b, (short)0, c, false, false);
}

__device__ __forceinline__ v16h pack16(h8_t lo, h8_t hi) {
  v16h r;
#pragma unroll
  for (int i = 0; i < 8; ++i) { r[i] = lo[i]; r[i+8] = hi[i]; }
  return r;
}

// A fragment, 16-bit 16x32: lane<16 holds row=lane, K {base+0..7, base+16..23};
// lane>=16 holds row=lane-16, K {base+8..15, base+24..31}. rowptr = this lane's row.
__device__ __forceinline__ v16h afrag(const _Float16* rowptr, int kbase, int half) {
  const int k0 = kbase + (half ? 8 : 0);
  return pack16(*(const h8_t*)(rowptr + k0), *(const h8_t*)(rowptr + k0 + 16));
}

// B fragment, 32x16: lane<16 holds col=lane, K {base+0..7, base+16..23}; lane>=16 col=lane-16, K {base+8..15, base+24..31}.
// WT is [128][kstride] f16 (transposed weights -> contiguous K runs).
__device__ __forceinline__ v16h bfrag(const _Float16* WT, int kstride, int ntile, int kbase, int lane) {
  const _Float16* p = WT + (ntile*16 + (lane & 15)) * kstride + kbase + ((lane >> 4) ? 8 : 0);
  return pack16(*(const h8_t*)p, *(const h8_t*)(p + 16));
}

// Load W[fin][128] (row-major fin x fout) transposed into LDS WT[128][kpad] f16, zero-padded K.
__device__ __forceinline__ void load_WT(_Float16* WT, const float* W, int fin, int kpad, int tid) {
  const int tot = 128 * kpad;
  for (int e = tid; e < tot; e += THREADS) {
    int n = e / kpad;
    int k = e - n * kpad;
    WT[e] = (_Float16)((k < fin) ? W[(size_t)k * 128 + n] : 0.0f);
  }
}

__global__ __launch_bounds__(THREADS)
void egcl_fused(Args A) {
  extern __shared__ char smem_raw[];
  Smem& S = *reinterpret_cast<Smem*>(smem_raw);
  const int b    = blockIdx.x;
  const int tid  = threadIdx.x;
  const int lane = tid & 31;
  const int wave = tid >> 5;
  const int half = lane >> 4;
  const int nloc = lane & 15;
  const int rb   = half * 8;        // D rows this lane holds: rb..rb+7
  const float* xb = A.x + (size_t)b * XROW;

  // ---------- Phase 0: inputs + encoder weights ----------
  for (int e = tid; e < NAG*32; e += THREADS) {
    int i = e >> 5, c = e & 31;
    S.featS[e] = (_Float16)((c < 17) ? xb[i*OBSF + c] : 0.0f);
  }
  if (tid < NAG) {
    int i = tid;
    bool v = false;
#pragma unroll
    for (int c = 0; c < 6; ++c) v = v || (xb[i*OBSF + c] != 0.0f);
    S.validf[i] = v ? 1.0f : 0.0f;
    S.posx[i] = xb[NAG*OBSF + 2*i];
    S.posy[i] = xb[NAG*OBSF + 2*i + 1];
  }
  load_WT(S.WT2, A.e0w, 17, 32, tid);     // enc0^T [128][32]
  load_WT(S.WT1, A.e1w, 128, 128, tid);   // enc1^T
  for (int e = tid; e < 128; e += THREADS) { S.biasA[e] = A.e0b[e]; S.biasB[e] = A.e1b[e]; }
  __syncthreads();

  // ---------- Encoder GEMM 1: feat(64x32) @ enc0 -> relu -> stageM ----------
  for (int job = wave; job < 32; job += NWAVE) {
    int mt = job & 3, nt = job >> 2;
    v8f acc = {};
    acc = wmma16(afrag(S.featS + (mt*16 + nloc)*32, 0, half),
                 bfrag(S.WT2, 32, nt, 0, lane), acc);
    int col = nt*16 + nloc;
#pragma unroll
    for (int j = 0; j < 8; ++j)
      S.stageM[(mt*16 + rb + j)*DIM + col] = (_Float16)fmaxf(acc[j] + S.biasA[col], 0.0f);
  }
  __syncthreads();
  // ---------- Encoder GEMM 2 -> relu -> h32/h16 ----------
  for (int job = wave; job < 32; job += NWAVE) {
    int mt = job & 3, nt = job >> 2;
    const _Float16* Ar = S.stageM + (mt*16 + nloc)*DIM;
    v8f acc = {};
#pragma unroll
    for (int c = 0; c < 4; ++c)
      acc = wmma16(afrag(Ar, c*32, half), bfrag(S.WT1, 128, nt, c*32, lane), acc);
    int col = nt*16 + nloc;
#pragma unroll
    for (int j = 0; j < 8; ++j) {
      int g = mt*16 + rb + j;
      float v = fmaxf(acc[j] + S.biasB[col], 0.0f);
      S.h32[g*DIM + col] = v;
      S.h16[g*DIM + col] = (_Float16)v;
    }
  }
  __syncthreads();

  // ---------- kNN (stable top-6 over squared distance; fixed thereafter) ----------
  if (tid < NAG) {
    const int i = tid;
    float bd[KNN]; int bi[KNN];
#pragma unroll
    for (int s = 0; s < KNN; ++s) { bd[s] = __builtin_inff(); bi[s] = 0; }
    const float pix = S.posx[i], piy = S.posy[i];
    const bool vi = (S.validf[i] != 0.0f);
    for (int j = 0; j < NAG; ++j) {
      float d2 = __builtin_inff();
      if (vi && S.validf[j] != 0.0f) {
        float dx = S.posx[j] - pix, dy = S.posy[j] - piy;
        d2 = dx*dx + dy*dy;
      }
      if (d2 < bd[KNN-1]) {            // strict < keeps earlier j on ties (stable argsort)
        int p2 = KNN-1;
        while (p2 > 0 && bd[p2-1] > d2) { bd[p2] = bd[p2-1]; bi[p2] = bi[p2-1]; --p2; }
        bd[p2] = d2; bi[p2] = j;
      }
    }
    float nn = 0.0f;
#pragma unroll
    for (int s = 0; s < KP; ++s) {
      int   nb2 = (s < KNN) ? bi[s] : 0;
      float mk  = (s < KNN && bd[s] <= 16.0f) ? 1.0f : 0.0f;   // sqrt(d2) <= 4
      S.nbrA[i*KP + s]  = nb2;
      S.maskA[i*KP + s] = mk;
      nn += mk;
    }
    S.nnumA[i] = nn;
  }
  __syncthreads();

  // ---------- 3 EGCL layers ----------
  for (int l = 0; l < 3; ++l) {
    const LayerP& Lp = A.L[l];
    // --- 3a: fe/fx weights + per-edge rel/dist from current pos + hm h-part ---
    load_WT(S.WT0, Lp.fe0w, 257, 288, tid);
    load_WT(S.WT1, Lp.fe1w, 128, 128, tid);
    load_WT(S.WT2, Lp.fx0w, 128, 128, tid);
    for (int e = tid; e < 128; e += THREADS) {
      S.biasA[e] = Lp.fe0b[e]; S.biasB[e] = Lp.fe1b[e]; S.biasC[e] = Lp.fx0b[e];
      S.fx1w[e]  = Lp.fx1w[e];
    }
    if (tid == 0) S.fx1b = Lp.fx1b[0];
    for (int e = tid; e < NAG*KP; e += THREADS) {
      int i = e >> 3;
      float rx = 0.0f, ry = 0.0f, dd = 0.0f;
      if (S.maskA[e] != 0.0f) {        // mask!=0 => both endpoints valid, pos finite
        int nb2 = S.nbrA[e];
        rx = S.posx[nb2] - S.posx[i];
        ry = S.posy[nb2] - S.posy[i];
        float sq = rx*rx + ry*ry;
        dd = (sq > 0.0f) ? sqrtf(sq) : 0.0f;   // safe_norm
      }
      S.relxA[e] = rx; S.relyA[e] = ry; S.distA[e] = dd;
    }
    for (int e = tid; e < NAG*DIM; e += THREADS)
      S.hm[(e >> 7)*(2*DIM) + (e & 127)] = S.h16[e];
    __syncthreads();

    // --- 3b: edge pipeline; 32 M-tiles of 16 edge-rows (2 agents/tile), 4 tiles/wave ---
    _Float16* T1 = S.stageS + wave*16*DIM;
    _Float16* TM = S.stageM + wave*16*DIM;
    for (int t = wave; t < 32; t += NWAVE) {
      const int a0 = 2*t, a1 = a0 + 1;
      const int myag = (nloc < 8) ? a0 : a1;   // this lane's row -> agent
      const int mysl = nloc & 7;
      const int mynb = S.nbrA[myag*KP + mysl];
      const float mydist = S.distA[myag*KP + mysl];
      const _Float16* Ah = S.h16 + myag*DIM;   // K 0..127  : h_i
      const _Float16* An = S.h16 + mynb*DIM;   // K 128..255: h_neigh (masked rows die later)
      v16h a8;                                  // K 256..287: [dist,0,...]
#pragma unroll
      for (int i = 0; i < 16; ++i) a8[i] = (_Float16)0.0f;
      if (half == 0) a8[0] = (_Float16)mydist;

      // fe0: 16x288 @ 288x128 -> silu -> T1
      for (int nt = 0; nt < 8; ++nt) {
        v8f acc = {};
#pragma unroll
        for (int c = 0; c < 4; ++c)
          acc = wmma16(afrag(Ah, c*32, half), bfrag(S.WT0, 288, nt, c*32, lane), acc);
#pragma unroll
        for (int c = 0; c < 4; ++c)
          acc = wmma16(afrag(An, c*32, half), bfrag(S.WT0, 288, nt, 128 + c*32, lane), acc);
        acc = wmma16(a8, bfrag(S.WT0, 288, nt, 256, lane), acc);
        int col = nt*16 + nloc;
#pragma unroll
        for (int j = 0; j < 8; ++j)
          T1[(rb + j)*DIM + col] = (_Float16)silu_f(acc[j] + S.biasA[col]);
      }
      // fe1 -> silu -> mask -> TM (this is m)
      for (int nt = 0; nt < 8; ++nt) {
        v8f acc = {};
        const _Float16* Ar = T1 + nloc*DIM;
#pragma unroll
        for (int c = 0; c < 4; ++c)
          acc = wmma16(afrag(Ar, c*32, half), bfrag(S.WT1, 128, nt, c*32, lane), acc);
        int col = nt*16 + nloc;
#pragma unroll
        for (int j = 0; j < 8; ++j) {
          int r = rb + j;
          int ag = (r < 8) ? a0 : a1;
          float mk = S.maskA[ag*KP + (r & 7)];
          TM[r*DIM + col] = (_Float16)(silu_f(acc[j] + S.biasB[col]) * mk);
        }
      }
      // fx0: m -> silu -> T1 (reuse)
      for (int nt = 0; nt < 8; ++nt) {
        v8f acc = {};
        const _Float16* Ar = TM + nloc*DIM;
#pragma unroll
        for (int c = 0; c < 4; ++c)
          acc = wmma16(afrag(Ar, c*32, half), bfrag(S.WT2, 128, nt, c*32, lane), acc);
        int col = nt*16 + nloc;
#pragma unroll
        for (int j = 0; j < 8; ++j)
          T1[(rb + j)*DIM + col] = (_Float16)silu_f(acc[j] + S.biasC[col]);
      }
      // fx1 (128 -> 1) + agg = sum(rel*w)/(nnum+1e-6); pos += agg
      float ax = 0.0f, ay = 0.0f;
      if (lane < 16) {
        const _Float16* mr = T1 + lane*DIM;
        float s = S.fx1b;
        for (int n2 = 0; n2 < DIM; ++n2) s += (float)mr[n2] * S.fx1w[n2];
        ax = S.relxA[myag*KP + mysl] * s;      // rel already masked (0)
        ay = S.relyA[myag*KP + mysl] * s;
      }
#pragma unroll
      for (int off = 1; off <= 4; off <<= 1) {
        ax += __shfl_xor(ax, off, 32);
        ay += __shfl_xor(ay, off, 32);
      }
      if (lane == 0) { float inv = 1.0f/(S.nnumA[a0] + 1e-6f); S.posx[a0] += ax*inv; S.posy[a0] += ay*inv; }
      if (lane == 8) { float inv = 1.0f/(S.nnumA[a1] + 1e-6f); S.posx[a1] += ax*inv; S.posy[a1] += ay*inv; }
      // m_i = sum_k m  -> hm[:, 128:256]
      for (int col = lane; col < DIM; col += 32) {
        float s0 = 0.0f, s1 = 0.0f;
#pragma unroll
        for (int r2 = 0; r2 < 8; ++r2) {
          s0 += (float)TM[r2*DIM + col];
          s1 += (float)TM[(8 + r2)*DIM + col];
        }
        S.hm[a0*(2*DIM) + DIM + col] = (_Float16)s0;
        S.hm[a1*(2*DIM) + DIM + col] = (_Float16)s1;
      }
    }
    __syncthreads();

    // --- 3c: fh weights (reuse WT0/WT1) ---
    load_WT(S.WT0, Lp.fh0w, 256, 256, tid);
    load_WT(S.WT1, Lp.fh1w, 128, 128, tid);
    for (int e = tid; e < 128; e += THREADS) { S.biasA[e] = Lp.fh0b[e]; S.biasB[e] = Lp.fh1b[e]; }
    __syncthreads();

    // --- 3d: fh0: hm(64x256) -> silu -> stageS[0:64][128] ---
    for (int job = wave; job < 32; job += NWAVE) {
      int mt = job & 3, nt = job >> 2;
      const _Float16* Ar = S.hm + (mt*16 + nloc)*(2*DIM);
      v8f acc = {};
#pragma unroll
      for (int c = 0; c < 8; ++c)
        acc = wmma16(afrag(Ar, c*32, half), bfrag(S.WT0, 256, nt, c*32, lane), acc);
      int col = nt*16 + nloc;
#pragma unroll
      for (int j = 0; j < 8; ++j)
        S.stageS[(mt*16 + rb + j)*DIM + col] = (_Float16)silu_f(acc[j] + S.biasA[col]);
    }
    __syncthreads();
    // fh1 -> residual into h32/h16
    for (int job = wave; job < 32; job += NWAVE) {
      int mt = job & 3, nt = job >> 2;
      const _Float16* Ar = S.stageS + (mt*16 + nloc)*DIM;
      v8f acc = {};
#pragma unroll
      for (int c = 0; c < 4; ++c)
        acc = wmma16(afrag(Ar, c*32, half), bfrag(S.WT1, 128, nt, c*32, lane), acc);
      int col = nt*16 + nloc;
#pragma unroll
      for (int j = 0; j < 8; ++j) {
        int g = mt*16 + rb + j;
        float nv = S.h32[g*DIM + col] + acc[j] + S.biasB[col];
        S.h32[g*DIM + col] = nv;
        S.h16[g*DIM + col] = (_Float16)nv;
      }
    }
    __syncthreads();
  }

  // ---------- Output ----------
  float* op = A.out + (size_t)b * NAG * DIM;
  for (int e = tid; e < NAG*DIM; e += THREADS) op[e] = S.h32[e];
}

extern "C" void kernel_launch(void* const* d_in, const int* in_sizes, int n_in,
                              void* d_out, int out_size, void* d_ws, size_t ws_size,
                              hipStream_t stream) {
  (void)d_ws; (void)ws_size; (void)out_size;
  Args A;
  // x is the 1024x1472 array (1,507,328 elems); weights are all <= 32,896 elems.
  int xi, p;
  if (in_sizes[0] > 100000) { xi = 0; p = 1; } else { xi = n_in - 1; p = 0; }
  A.x = (const float*)d_in[xi];
  A.e0w = (const float*)d_in[p+0]; A.e0b = (const float*)d_in[p+1];
  A.e1w = (const float*)d_in[p+2]; A.e1b = (const float*)d_in[p+3];
  p += 4;
  for (int l = 0; l < 3; ++l) {
    LayerP& L = A.L[l];
    L.fe0w = (const float*)d_in[p+0]; L.fe0b = (const float*)d_in[p+1];
    L.fe1w = (const float*)d_in[p+2]; L.fe1b = (const float*)d_in[p+3];
    if (in_sizes[p+4] == 32768) {          // jax-sorted dict keys: fe, fh, fx
      L.fh0w = (const float*)d_in[p+4];  L.fh0b = (const float*)d_in[p+5];
      L.fh1w = (const float*)d_in[p+6];  L.fh1b = (const float*)d_in[p+7];
      L.fx0w = (const float*)d_in[p+8];  L.fx0b = (const float*)d_in[p+9];
      L.fx1w = (const float*)d_in[p+10]; L.fx1b = (const float*)d_in[p+11];
    } else {                               // insertion order: fe, fx, fh
      L.fx0w = (const float*)d_in[p+4];  L.fx0b = (const float*)d_in[p+5];
      L.fx1w = (const float*)d_in[p+6];  L.fx1b = (const float*)d_in[p+7];
      L.fh0w = (const float*)d_in[p+8];  L.fh0b = (const float*)d_in[p+9];
      L.fh1w = (const float*)d_in[p+10]; L.fh1b = (const float*)d_in[p+11];
    }
    p += 12;
  }
  A.out = (float*)d_out;

  const size_t smem = sizeof(Smem);   // ~297 KB; CDNA5 WGP allows up to 320 KB
  hipFuncSetAttribute((const void*)egcl_fused,
                      hipFuncAttributeMaxDynamicSharedMemorySize, (int)smem);
  egcl_fused<<<NBATCH, THREADS, smem, stream>>>(A);
}
